// Unit_60687887892937
// MI455X (gfx1250) — matrix operand order repaired and checked by reading, not compiled, not run on hardware
//
#include <hip/hip_runtime.h>
#include <hip/hip_bf16.h>

// ---------------------------------------------------------------------------
// 2s-AGCN block on MI455X (gfx1250, wave32).  All GEMM-shaped work lowered to
// v_wmma_f32_16x16x32_f16 (f16 operands, fp32 accumulate).  Softmax / BN /
// residuals in fp32.  Compute-bound (~30 GFLOP), intermediates L2-resident.
// This revision adds: software pipelining of fragment loads, dual-N-tile
// accumulators per A fragment, and async global->LDS staging (ASYNCcnt path)
// of the temporal-conv weight tile.
// ---------------------------------------------------------------------------

typedef __attribute__((ext_vector_type(16))) _Float16 v16h;
typedef __attribute__((ext_vector_type(8)))  _Float16 v8h;
typedef __attribute__((ext_vector_type(8)))  float    v8f;

constexpr int Nn = 32, Cc = 64, Tt = 300, Vv = 25, NSs = 3, INTERi = 16, Kk = 9;
constexpr int TV   = Tt * Vv;       // 7500
constexpr int CTV  = Cc * TV;       // 480000
constexpr int NTV  = Nn * TV;       // 240000
constexpr int NT16 = NTV / 16;      // 15000 column tiles
constexpr int T32  = Tt * 32;       // 9600
constexpr int NT32 = Nn * T32;      // 307200 (P columns per channel)
constexpr int TP   = Tt + 8;        // 308 (temporal halo)
constexpr int KW   = Kk * Cc;       // 576 (conv GEMM K)
constexpr int KATT = INTERi * Tt;   // 4800 (attention GEMM K)
constexpr float EPSf = 1e-5f;

// ---- workspace layout (bytes, all 256-aligned) ----------------------------
constexpr size_t OFF_XH   = 0;                          // xh   [NTV][64] f16       30,720,000
constexpr size_t OFF_EMB  = 30720000;                   // emb  58,982,400  (P aliases: 39,321,600)
constexpr size_t OFF_WAB  = 89702400;                   //       12,288
constexpr size_t OFF_WD   = 89714688;                   //       24,576
constexpr size_t OFF_WT   = 89739264;                   //       73,728
constexpr size_t OFF_ATTN = 89812992;                   //      196,608
constexpr size_t OFF_YG   = 90009600;                   // ygcn f32 (yconv aliases) 61,440,000
constexpr size_t OFF_Y1H  = 151449600;                  // y1h f16  31,539,200
constexpr size_t OFF_ST   = 182988800;                  // stats 1,024

#define WMMA_F16(a, b, c) \
  __builtin_amdgcn_wmma_f32_16x16x32_f16(false, (a), false, (b), (short)0, (c), false, false)

// A fragment 16x32 f16: per-lane row pointer (row m = lane%16, at k0).
// Layout: lanes<16 -> K halves {0..15}; lanes>=16 -> {16..31} (matches fragB).
__device__ inline v16h fragA(const _Float16* rowp, int lane) {
  int kh = (lane >> 4) * 16;
  v8h lo = *(const v8h*)(rowp + kh);
  v8h hi = *(const v8h*)(rowp + kh + 8);
  v16h o;
#pragma unroll
  for (int j = 0; j < 8; ++j) { o[j] = lo[j]; o[j + 8] = hi[j]; }
  return o;
}
// B fragment 32x16 f16: per-lane column pointer already offset by
// k0 + (lane>>4)*16; 16 contiguous K halves.
__device__ inline v16h fragB(const _Float16* colp) {
  v8h lo = *(const v8h*)(colp);
  v8h hi = *(const v8h*)(colp + 8);
  v16h o;
#pragma unroll
  for (int j = 0; j < 8; ++j) { o[j] = lo[j]; o[j + 8] = hi[j]; }
  return o;
}

// ---------------------------------------------------------------------------
__global__ void k_zero16(uint4* p, int n128) {
  int i = blockIdx.x * blockDim.x + threadIdx.x;
  if (i < n128) p[i] = make_uint4(0u, 0u, 0u, 0u);
}

// xh[j][c] = x[n][c][t][v]  (channels contiguous -> contiguous B fragments)
__global__ void k_pack_x(const float* __restrict__ x, _Float16* __restrict__ xh) {
  int j = blockIdx.x * blockDim.x + threadIdx.x;
  if (j >= NTV) return;
  int n = j / TV, r = j % TV;
  const float* xp = x + (size_t)n * CTV + r;
  _Float16* o = xh + (size_t)j * Cc;
#pragma unroll 4
  for (int c = 0; c < Cc; ++c) o[c] = (_Float16)xp[(size_t)c * TV];
}

// Pack weights to f16 GEMM layouts.
__global__ void k_prep_w(const float* __restrict__ Wa, const float* __restrict__ Wb,
                         const float* __restrict__ Wd, const float* __restrict__ Wt,
                         _Float16* __restrict__ wab, _Float16* __restrict__ wdh,
                         _Float16* __restrict__ wth) {
  int t0 = blockIdx.x * blockDim.x + threadIdx.x;
  int stride = blockDim.x * gridDim.x;
  for (int idx = t0; idx < 96 * 64; idx += stride) {        // rows: ab*48 + i*16 + o
    int row = idx >> 6, c = idx & 63;
    int ab = row / 48, io = row % 48;
    wab[idx] = (_Float16)(ab ? Wb[io * Cc + c] : Wa[io * Cc + c]);
  }
  for (int idx = t0; idx < NSs * Cc * Cc; idx += stride)    // [i][o][c] row-major
    wdh[idx] = (_Float16)Wd[idx];
  for (int idx = t0; idx < Cc * KW; idx += stride) {        // [c][kk*64+ci]
    int c = idx / KW, rem = idx % KW, kk = rem >> 6, ci = rem & 63;
    wth[idx] = (_Float16)Wt[((size_t)(c * Cc + ci)) * Kk + kk];
  }
}

// emb = Wab x (+ba/bb); each wave: 1 M-tile x 2 N-tiles (A fragment reused),
// scattered to emb[n][v(32)][i][ab][o*T+t] f16.
__global__ void k_emb(const _Float16* __restrict__ wab, const _Float16* __restrict__ xh,
                      const float* __restrict__ ba, const float* __restrict__ bb,
                      _Float16* __restrict__ emb) {
  int w = blockIdx.x * (blockDim.x >> 5) + (threadIdx.x >> 5);
  int lane = threadIdx.x & 31;
  int mt = w / (NT16 / 2), np = w % (NT16 / 2);    // 6 x 7500 (tile pairs)
  int m0 = mt * 16;
  int j0 = np * 32 + (lane & 15);
  int j1 = j0 + 16;
  const _Float16* rowb  = wab + (size_t)(m0 + (lane & 15)) * Cc;
  const _Float16* colb0 = xh + (size_t)j0 * Cc + ((lane >> 4) * 16);
  const _Float16* colb1 = xh + (size_t)j1 * Cc + ((lane >> 4) * 16);
  v8f acc0 = {}, acc1 = {};
#pragma unroll
  for (int k0 = 0; k0 < 64; k0 += 32) {
    v16h a = fragA(rowb + k0, lane);
    acc0 = WMMA_F16(a, fragB(colb0 + k0), acc0);
    acc1 = WMMA_F16(a, fragB(colb1 + k0), acc1);
  }
#pragma unroll
  for (int half = 0; half < 2; ++half) {
    int j = half ? j1 : j0;
    const v8f& acc = half ? acc1 : acc0;
    int n = j / TV, r = j % TV, tt = r / Vv, vv = r % Vv;
#pragma unroll
    for (int r8 = 0; r8 < 8; ++r8) {
      int m = m0 + r8 + 8 * (lane >> 4);
      int ab = m / 48, io = m % 48;
      float bias = ab ? bb[io] : ba[io];
      size_t idx = ((((size_t)n * 32 + vv) * NSs + (io >> 4)) * 2 + ab) * (size_t)KATT
                   + (io & 15) * Tt + tt;
      emb[idx] = (_Float16)(acc[r8] + bias);
    }
  }
}

// Attention: S = A1 A2 over K=4800 (2-stage pipelined), softmax over u, +Au;
// write attnT[i][n][v][u] f16 (u-contiguous; pads zero).
__global__ void k_attn(const _Float16* __restrict__ emb, const float* __restrict__ Aadj,
                       const float* __restrict__ PA, _Float16* __restrict__ attn) {
  int bi = blockIdx.x;                      // NS*N blocks
  int i = bi / Nn, n = bi % Nn;
  int lane = threadIdx.x & 31, wv = threadIdx.x >> 5;   // 2 waves: v-tiles
  int v0 = wv * 16;
  const size_t ustride = (size_t)NSs * 2 * KATT;        // 28800
  size_t baseA = (((size_t)n * 32 * NSs + i) * 2 + 0) * (size_t)KATT;
  size_t baseB = (((size_t)n * 32 * NSs + i) * 2 + 1) * (size_t)KATT;
  int mA = lane & 15;
  const _Float16* colp  = emb + baseB + (size_t)(v0 + (lane & 15)) * ustride + ((lane >> 4) * 16);
  const _Float16* arow0 = emb + baseA + (size_t)(0 + mA) * ustride;
  const _Float16* arow1 = emb + baseA + (size_t)(16 + mA) * ustride;
  v8f acc0 = {}, acc1 = {};
  v16h b  = fragB(colp);
  v16h a0 = fragA(arow0, lane);
  v16h a1 = fragA(arow1, lane);
  for (int k0 = 0; k0 < KATT; k0 += 32) {
    int kn = (k0 + 32 < KATT) ? k0 + 32 : 0;          // wrap-safe prefetch addr
    v16h bn  = fragB(colp + kn);
    v16h an0 = fragA(arow0 + kn, lane);
    v16h an1 = fragA(arow1 + kn, lane);
    acc0 = WMMA_F16(a0, b, acc0);
    acc1 = WMMA_F16(a1, b, acc1);
    b = bn; a0 = an0; a1 = an1;
  }
  __shared__ float S[32][33];
  int vcol = v0 + (lane & 15);
#pragma unroll
  for (int r8 = 0; r8 < 8; ++r8) {
    S[r8 + 8 * (lane >> 4)][vcol]      = acc0[r8];
    S[16 + r8 + 8 * (lane >> 4)][vcol] = acc1[r8];
  }
  __syncthreads();
  int v = threadIdx.x;
  if (v < 32) {
    const float scale = 1.0f / (float)KATT;
    float out[32];
#pragma unroll
    for (int u = 0; u < 32; ++u) out[u] = 0.f;
    if (v < Vv) {
      float mx = -1e30f;
      for (int u = 0; u < Vv; ++u) mx = fmaxf(mx, S[u][v] * scale);
      float sum = 0.f;
      for (int u = 0; u < Vv; ++u) { float e = __expf(S[u][v] * scale - mx); out[u] = e; sum += e; }
      float inv = 1.f / sum;
      for (int u = 0; u < Vv; ++u)
        out[u] = out[u] * inv + Aadj[(i * Vv + u) * Vv + v] + PA[(i * Vv + u) * Vv + v];
    }
    _Float16* dst = attn + (((size_t)i * Nn + n) * 32 + v) * 32;
    for (int u = 0; u < 32; ++u)
      dst[u] = (_Float16)((v < Vv && u < Vv) ? out[u] : 0.f);
  }
}

// P = Wd_i * x  (K=64), 1 M-tile x 2 N-tiles per wave.  P[c][n][t][u(32)],
// u-pads killed later by zero attn rows.
__global__ void k_zg1(const _Float16* __restrict__ wdh, const _Float16* __restrict__ xh,
                      _Float16* __restrict__ P, int i) {
  int w = blockIdx.x * (blockDim.x >> 5) + (threadIdx.x >> 5);
  int lane = threadIdx.x & 31;
  int mt = w / (NT16 / 2), np = w % (NT16 / 2);    // 4 x 7500
  int m0 = mt * 16;
  const _Float16* rowb = wdh + (size_t)i * Cc * Cc + (size_t)(m0 + (lane & 15)) * Cc;
  int j0 = np * 32 + (lane & 15);
  int j1 = j0 + 16;
  const _Float16* colb0 = xh + (size_t)j0 * Cc + ((lane >> 4) * 16);
  const _Float16* colb1 = xh + (size_t)j1 * Cc + ((lane >> 4) * 16);
  v8f acc0 = {}, acc1 = {};
#pragma unroll
  for (int k0 = 0; k0 < 64; k0 += 32) {
    v16h a = fragA(rowb + k0, lane);
    acc0 = WMMA_F16(a, fragB(colb0 + k0), acc0);
    acc1 = WMMA_F16(a, fragB(colb1 + k0), acc1);
  }
#pragma unroll
  for (int half = 0; half < 2; ++half) {
    int j = half ? j1 : j0;
    const v8f& acc = half ? acc1 : acc0;
    int n = j / TV, r = j % TV, tt = r / Vv, uu = r % Vv;
    size_t base = (size_t)n * T32 + tt * 32 + uu;
#pragma unroll
    for (int r8 = 0; r8 < 8; ++r8) {
      int c = m0 + r8 + 8 * (lane >> 4);
      P[(size_t)c * NT32 + base] = (_Float16)acc[r8];
    }
  }
}

// y_gcn(+)= P @ attn_i  (K=32 single WMMA step), +bd_i.
__global__ void k_zg2(const _Float16* __restrict__ P, const _Float16* __restrict__ attn,
                      const float* __restrict__ bd, float* __restrict__ yg, int i) {
  int w = blockIdx.x * (blockDim.x >> 5) + (threadIdx.x >> 5);
  int lane = threadIdx.x & 31;
  int n = w / 2400, rem = w % 2400, mt = rem >> 1, vt = rem & 1;
  int m = mt * 16 + (lane & 15);                   // A row = (c,t)
  int c = m / Tt, tt = m % Tt;
  const _Float16* arow = P + (size_t)c * NT32 + (size_t)n * T32 + tt * 32;
  int v = vt * 16 + (lane & 15);
  const _Float16* colp = attn + (((size_t)i * Nn + n) * 32 + v) * 32 + ((lane >> 4) * 16);
  v8f acc = {};
  acc = WMMA_F16(fragA(arow, lane), fragB(colp), acc);
  if (v < Vv) {
#pragma unroll
    for (int r8 = 0; r8 < 8; ++r8) {
      int mm = mt * 16 + r8 + 8 * (lane >> 4);
      int cc = mm / Tt, t2 = mm % Tt;
      size_t idx = (size_t)n * CTV + (size_t)cc * TV + t2 * Vv + v;
      float val = acc[r8] + bd[i * Cc + cc];
      if (i) val += yg[idx];
      yg[idx] = val;
    }
  }
}

// Per-channel sum / sumsq for training-mode BN.
__global__ void k_bnstats(const float* __restrict__ y, float* __restrict__ stats) {
  int c = blockIdx.x;
  float s = 0.f, ss = 0.f;
  for (int j = threadIdx.x; j < Nn * TV; j += blockDim.x) {
    int n = j / TV, r = j % TV;
    float v = y[(size_t)n * CTV + (size_t)c * TV + r];
    s += v; ss += v * v;
  }
  __shared__ float as[256], bs[256];
  as[threadIdx.x] = s; bs[threadIdx.x] = ss; __syncthreads();
  for (int k = 128; k > 0; k >>= 1) {
    if (threadIdx.x < k) { as[threadIdx.x] += as[threadIdx.x + k]; bs[threadIdx.x] += bs[threadIdx.x + k]; }
    __syncthreads();
  }
  if (threadIdx.x == 0) { stats[c] = as[0]; stats[64 + c] = bs[0]; }
}

// y1 = relu(bn1(y_gcn) + x) -> f16 [n][t+4][v][c] (channel-contiguous, halo zeroed).
__global__ void k_post1(const float* __restrict__ yg, const float* __restrict__ x,
                        const float* __restrict__ stats, const float* __restrict__ g1,
                        const float* __restrict__ b1, _Float16* __restrict__ y1h) {
  size_t idx = (size_t)blockIdx.x * blockDim.x + threadIdx.x;
  if (idx >= (size_t)Nn * CTV) return;
  int n = (int)(idx / CTV), rem = (int)(idx % CTV);
  int c = rem / TV, r = rem % TV, tt = r / Vv, vv = r % Vv;
  const float cnt = (float)(Nn * TV);
  float mean = stats[c] / cnt;
  float var  = stats[64 + c] / cnt - mean * mean;
  float val  = (yg[idx] - mean) * rsqrtf(var + EPSf) * g1[c] + b1[c] + x[idx];
  val = fmaxf(val, 0.f);
  y1h[(((size_t)n * TP + tt + 4) * Vv + vv) * Cc + c] = (_Float16)val;
}

// Temporal conv as implicit-im2col GEMM, K = 576 (18 WMMA steps).
// The 16x576 f16 weight tile is staged to LDS once per block with the gfx1250
// async global->LDS copy (ASYNCcnt), then each of 4 waves runs 2 N-tiles with
// pipelined B fragment loads.
__global__ void k_conv(const _Float16* __restrict__ wth, const _Float16* __restrict__ y1h,
                       const float* __restrict__ bt, float* __restrict__ yc) {
  __shared__ alignas(16) _Float16 Alds[16 * KW];          // 18,432 B
  int mt = blockIdx.x / 1875;                              // 4 M-tiles
  int nb = blockIdx.x % 1875;                              // 8 N-tiles per block
  int m0 = mt * 16;
  // Stage A tile: 1152 x b128 async copies across 128 threads (9 each).
  for (int e = threadIdx.x; e < (16 * KW) / 8; e += blockDim.x) {
    unsigned lo = (unsigned)(size_t)(&Alds[e * 8]);       // LDS byte offset (addr[31:0])
    const _Float16* gp = wth + (size_t)m0 * KW + (size_t)e * 8;
    asm volatile("global_load_async_to_lds_b128 %0, %1, off"
                 :: "v"(lo), "v"(gp) : "memory");
  }
  asm volatile("s_wait_asynccnt 0x0" ::: "memory");
  __syncthreads();

  int wv = threadIdx.x >> 5, lane = threadIdx.x & 31;
  int nt0 = nb * 8 + wv * 2;                               // two adjacent N-tiles
  int j0 = nt0 * 16 + (lane & 15);
  int j1 = j0 + 16;
  int n0 = j0 / TV, r0 = j0 % TV, tt0 = r0 / Vv, vv0 = r0 % Vv;
  int n1 = j1 / TV, r1 = j1 % TV, tt1 = r1 / Vv, vv1 = r1 % Vv;
  size_t base0 = ((size_t)n0 * TP + tt0) * (Vv * Cc) + (size_t)vv0 * Cc;
  size_t base1 = ((size_t)n1 * TP + tt1) * (Vv * Cc) + (size_t)vv1 * Cc;
  const _Float16* arow = Alds + (size_t)(lane & 15) * KW;

  auto bload = [&](int k0, size_t base) -> v16h {
    int kb = k0 + ((lane >> 4) << 4);                      // 16-chunk within one tap
    int kk = kb >> 6, cb = kb & 63;
    return fragB(y1h + base + (size_t)kk * (Vv * Cc) + cb);
  };

  v8f acc0 = {}, acc1 = {};
  v16h b0 = bload(0, base0);
  v16h b1 = bload(0, base1);
  for (int k0 = 0; k0 < KW; k0 += 32) {
    int kn = (k0 + 32 < KW) ? k0 + 32 : 0;                 // wrap-safe prefetch addr
    v16h nb0 = bload(kn, base0);
    v16h nb1 = bload(kn, base1);
    v16h a = fragA(arow + k0, lane);                       // LDS reads
    acc0 = WMMA_F16(a, b0, acc0);
    acc1 = WMMA_F16(a, b1, acc1);
    b0 = nb0; b1 = nb1;
  }
#pragma unroll
  for (int half = 0; half < 2; ++half) {
    const v8f& acc = half ? acc1 : acc0;
    int n = half ? n1 : n0, tt = half ? tt1 : tt0, vv = half ? vv1 : vv0;
#pragma unroll
    for (int r8 = 0; r8 < 8; ++r8) {
      int c = m0 + r8 + 8 * (lane >> 4);
      yc[(size_t)n * CTV + (size_t)c * TV + tt * Vv + vv] = acc[r8] + bt[c];
    }
  }
}

// out = relu(bn2(yconv) + x)  (fp32 output)
__global__ void k_post2(const float* __restrict__ yc, const float* __restrict__ x,
                        const float* __restrict__ stats, const float* __restrict__ g2,
                        const float* __restrict__ b2, float* __restrict__ out) {
  size_t idx = (size_t)blockIdx.x * blockDim.x + threadIdx.x;
  if (idx >= (size_t)Nn * CTV) return;
  int c = (int)((idx / TV) % Cc);
  const float cnt = (float)(Nn * TV);
  float mean = stats[c] / cnt;
  float var  = stats[64 + c] / cnt - mean * mean;
  float val  = (yc[idx] - mean) * rsqrtf(var + EPSf) * g2[c] + b2[c] + x[idx];
  out[idx] = fmaxf(val, 0.f);
}

// ---------------------------------------------------------------------------
extern "C" void kernel_launch(void* const* d_in, const int* in_sizes, int n_in,
                              void* d_out, int out_size, void* d_ws, size_t ws_size,
                              hipStream_t stream) {
  const float* x  = (const float*)d_in[0];
  const float* A  = (const float*)d_in[1];
  const float* PA = (const float*)d_in[2];
  const float* Wa = (const float*)d_in[3];
  const float* ba = (const float*)d_in[4];
  const float* Wb = (const float*)d_in[5];
  const float* bb = (const float*)d_in[6];
  const float* Wd = (const float*)d_in[7];
  const float* bd = (const float*)d_in[8];
  const float* g1 = (const float*)d_in[9];
  const float* b1 = (const float*)d_in[10];
  const float* Wt = (const float*)d_in[11];
  const float* bt = (const float*)d_in[12];
  const float* g2 = (const float*)d_in[13];
  const float* b2 = (const float*)d_in[14];
  float* out = (float*)d_out;

  char* ws = (char*)d_ws;
  _Float16* xh   = (_Float16*)(ws + OFF_XH);
  _Float16* emb  = (_Float16*)(ws + OFF_EMB);
  _Float16* Pbuf = (_Float16*)(ws + OFF_EMB);   // aliases emb (emb dead by then)
  _Float16* wab  = (_Float16*)(ws + OFF_WAB);
  _Float16* wdh  = (_Float16*)(ws + OFF_WD);
  _Float16* wth  = (_Float16*)(ws + OFF_WT);
  _Float16* attn = (_Float16*)(ws + OFF_ATTN);
  float*    yg   = (float*)   (ws + OFF_YG);
  float*    yc   = (float*)   (ws + OFF_YG);    // aliases ygcn (dead after post1)
  _Float16* y1h  = (_Float16*)(ws + OFF_Y1H);
  float*    st1  = (float*)   (ws + OFF_ST);
  float*    st2  = (float*)   (ws + OFF_ST) + 128;

  // zero the temporal-conv halo buffer (31.5 MB as uint4)
  {
    int n128 = (Nn * TP * Vv * Cc * 2) / 16;    // 1,971,200
    k_zero16<<<(n128 + 255) / 256, 256, 0, stream>>>((uint4*)y1h, n128);
  }
  k_pack_x<<<(NTV + 255) / 256, 256, 0, stream>>>(x, xh);
  k_prep_w<<<64, 256, 0, stream>>>(Wa, Wb, Wd, Wt, wab, wdh, wth);

  // embeddings: 6 M-tiles x 7500 N-tile pairs, 2 waves/block
  k_emb<<<22500, 64, 0, stream>>>(wab, xh, ba, bb, emb);
  // attention + softmax: one block per (i, n)
  k_attn<<<NSs * Nn, 64, 0, stream>>>(emb, A, PA, attn);

  for (int i = 0; i < NSs; ++i) {
    k_zg1<<<15000, 64, 0, stream>>>(wdh, xh, Pbuf, i);      // 4 x 7500 tile pairs
    k_zg2<<<38400, 64, 0, stream>>>(Pbuf, attn, bd, yg, i); // 32 x 1200 x 2 tiles
  }

  k_bnstats<<<Cc, 256, 0, stream>>>(yg, st1);
  k_post1<<<(Nn * CTV + 255) / 256, 256, 0, stream>>>(yg, x, st1, g1, b1, y1h);

  // temporal conv: 4 M-tiles x 1875 blocks, 128 threads (4 waves x 2 N-tiles)
  k_conv<<<7500, 128, 0, stream>>>(wth, y1h, bt, yc);

  k_bnstats<<<Cc, 256, 0, stream>>>(yc, st2);
  k_post2<<<(Nn * CTV + 255) / 256, 256, 0, stream>>>(yc, x, st2, g2, b2, out);

  (void)in_sizes; (void)n_in; (void)out_size; (void)ws_size;
}
